// DJMGNN_74285754352147
// MI455X (gfx1250) — compile-verified, matrix operand order checked by reading, not run on hardware
//
#include <hip/hip_runtime.h>
#include <hip/hip_bf16.h>

typedef __attribute__((ext_vector_type(16))) _Float16 v16h;
typedef __attribute__((ext_vector_type(8)))  float    v8f;

#define GN_EPS 1e-5f

// ---------------------------------------------------------------------------
// Fused edge kernel: for a tile of 16 edges per wave (4 waves / block):
//   ew = relu(edge_attr[e] @ Wmlp + b)   via v_wmma_f32_16x16x32_f16
//   msg[e] = h[src[e]] @ ew              fused per 16-col WMMA tile (VALU)
//   atomicAdd into agg[dst[e]]           (segment_sum)
// Wmlp (32x1024) is staged in LDS as f16, pre-swizzled into the exact B-operand
// lane layout (lane<16: K 0-7,16-23; lane>=16: K 8-15,24-31) so each lane loads
// its 16 halfs as one contiguous 32B ds read. Two passes of 32 col-chunks keep
// static LDS < 64KB.
// ---------------------------------------------------------------------------
__global__ __launch_bounds__(128) void edge_conv_kernel(
    const float* __restrict__ edge_attr,   // [E,32]
    const int*   __restrict__ edge_index,  // [2,E]
    const float* __restrict__ h,           // [N,32]
    const float* __restrict__ Wmlp,        // [32,1024]  (layer slice)
    const float* __restrict__ bmlp,        // [1024]
    float*       __restrict__ agg,         // [N,32] atomic accumulate
    int E_, int N_)
{
    (void)N_;
    __shared__ __align__(32) _Float16 Wsw[32 * 512];   // 32 chunks / pass, 32KB
    __shared__ float btile[1024];                      // 4KB
    __shared__ float htile[4][16 * 32];                // 8KB
    __shared__ int   srcs[4][16], dsts[4][16];

    const int tid  = threadIdx.x;
    const int lane = tid & 31;
    const int w    = tid >> 5;
    const int ebase = (blockIdx.x * 4 + w) * 16;

    if (lane < 16) {
        int e = ebase + lane; if (e >= E_) e = E_ - 1;
        srcs[w][lane] = edge_index[e];
        dsts[w][lane] = edge_index[E_ + e];
    }
    for (int s = tid; s < 1024; s += 128) btile[s] = bmlp[s];
    __syncthreads();

    // gather h[src] tile: 16 edges x 32 feats
    for (int r = 0; r < 16; ++r)
        htile[w][r * 32 + lane] = h[(size_t)srcs[w][r] * 32 + lane];

    // A operand: edge_attr tile, 16x32 f16, hardware lane layout
    int erow   = ebase + (lane & 15);
    int eclamp = erow < E_ ? erow : E_ - 1;
    const float* ap = edge_attr + (size_t)eclamp * 32;
    const int kb0 = (lane < 16) ? 0  : 8;
    const int kb1 = (lane < 16) ? 16 : 24;
    v16h A;
#pragma unroll
    for (int j = 0; j < 8; ++j) {
        A[j]     = (_Float16)ap[kb0 + j];
        A[8 + j] = (_Float16)ap[kb1 + j];
    }

    float accL[8], accH[8];
#pragma unroll
    for (int v = 0; v < 8; ++v) { accL[v] = 0.f; accH[v] = 0.f; }
    const int rbase = (lane < 16) ? 0 : 8;

    for (int pass = 0; pass < 2; ++pass) {
        __syncthreads();
        // stage 32 swizzled B-chunks of Wmlp (cols pass*512 .. pass*512+511)
        for (int s = tid; s < 32 * 512; s += 128) {
            int cl = s >> 9, l = (s >> 4) & 31, e2 = s & 15;
            int chunk = pass * 32 + cl;
            int col = ((chunk >> 1) << 5) + ((chunk & 1) << 4) + (l & 15);
            int kl  = (l < 16) ? (e2 < 8 ? e2 : e2 + 8)
                               : (e2 < 8 ? e2 + 8 : e2 + 16);
            Wsw[s] = (_Float16)Wmlp[(size_t)kl * 1024 + col];
        }
        __syncthreads();

        for (int il = 0; il < 16; ++il) {
            int i = pass * 16 + il;           // feature index of ew row
#pragma unroll
            for (int half = 0; half < 2; ++half) {
                int cl  = il * 2 + half;
                int col = (i << 5) + (half << 4) + (lane & 15);
                v16h B = *(const v16h*)&Wsw[(cl * 32 + lane) * 16];
                float bv = btile[col];
                v8f c;
#pragma unroll
                for (int v = 0; v < 8; ++v) c[v] = bv;   // bias folded into C
                c = __builtin_amdgcn_wmma_f32_16x16x32_f16(
                        false, A, false, B, (short)0, c, false, false);
#pragma unroll
                for (int v = 0; v < 8; ++v) {
                    float hv  = htile[w][(rbase + v) * 32 + i];
                    float ewv = c[v] > 0.f ? c[v] : 0.f;   // relu
                    if (half == 0) accL[v] += hv * ewv;
                    else           accH[v] += hv * ewv;
                }
            }
        }
    }

    // scatter-add messages to agg[dst]
    const int colL = lane & 15, colH = 16 + colL;
#pragma unroll
    for (int v = 0; v < 8; ++v) {
        int er = rbase + v;
        int e  = ebase + er;
        if (e < E_) {
            int d = dsts[w][er];
            atomicAdd(&agg[(size_t)d * 32 + colL], accL[v]);
            atomicAdd(&agg[(size_t)d * 32 + colH], accH[v]);
        }
    }
}

// ---------------------------------------------------------------------------
// Node-side WMMA GEMM: Y[N,Nout] = opt_relu( X[N,K] @ W[K,Nout] + bias + addC )
// K = NKC*32, Nout = NNC*16. If XB != nullptr, X = concat([XA,XB]) (each N x 32).
// ---------------------------------------------------------------------------
template <int NKC, int NNC>
__global__ __launch_bounds__(128) void node_gemm_kernel(
    const float* __restrict__ XA, const float* __restrict__ XB,
    const float* __restrict__ W,  const float* __restrict__ bias,
    const float* __restrict__ addC, float* __restrict__ Y,
    int Nrows, int doRelu)
{
    constexpr int Kdim = NKC * 32;
    constexpr int Nout = NNC * 16;
    constexpr int SLOTS = NKC * NNC * 512;
    __shared__ __align__(32) _Float16 Wsw[SLOTS];

    const int tid = threadIdx.x, lane = tid & 31, w = tid >> 5;

    for (int s = tid; s < SLOTS; s += 128) {
        int chunkIdx = s >> 9, l = (s >> 4) & 31, e2 = s & 15;
        int kc = chunkIdx / NNC, nc = chunkIdx % NNC;
        int col = nc * 16 + (l & 15);
        int kl  = (l < 16) ? (e2 < 8 ? e2 : e2 + 8)
                           : (e2 < 8 ? e2 + 8 : e2 + 16);
        Wsw[s] = (_Float16)W[(size_t)(kc * 32 + kl) * Nout + col];
    }
    __syncthreads();

    const int rowbase = (blockIdx.x * 4 + w) * 16;
    const int myrow   = rowbase + (lane & 15);
    const int rclamp  = myrow < Nrows ? myrow : Nrows - 1;
    const int colb    = lane & 15;

    v8f acc[NNC];
#pragma unroll
    for (int nc = 0; nc < NNC; ++nc) {
        float bv = bias[nc * 16 + colb];
#pragma unroll
        for (int v = 0; v < 8; ++v) acc[nc][v] = bv;
    }

#pragma unroll
    for (int kc = 0; kc < NKC; ++kc) {
        v16h A;
        const int kb0 = kc * 32 + ((lane < 16) ? 0  : 8);
        const int kb1 = kc * 32 + ((lane < 16) ? 16 : 24);
#pragma unroll
        for (int j = 0; j < 8; ++j) {
            int k0 = kb0 + j, k1 = kb1 + j;
            float f0, f1;
            if (XB) {  // concat mode: K==64, halves from XA / XB
                f0 = (k0 < 32) ? XA[(size_t)rclamp * 32 + k0]
                               : XB[(size_t)rclamp * 32 + (k0 - 32)];
                f1 = (k1 < 32) ? XA[(size_t)rclamp * 32 + k1]
                               : XB[(size_t)rclamp * 32 + (k1 - 32)];
            } else {
                f0 = XA[(size_t)rclamp * Kdim + k0];
                f1 = XA[(size_t)rclamp * Kdim + k1];
            }
            A[j]     = (_Float16)f0;
            A[8 + j] = (_Float16)f1;
        }
#pragma unroll
        for (int nc = 0; nc < NNC; ++nc) {
            v16h B = *(const v16h*)&Wsw[((kc * NNC + nc) * 32 + lane) * 16];
            acc[nc] = __builtin_amdgcn_wmma_f32_16x16x32_f16(
                          false, A, false, B, (short)0, acc[nc], false, false);
        }
    }

    const int rb = (lane < 16) ? 0 : 8;
#pragma unroll
    for (int nc = 0; nc < NNC; ++nc) {
        int col = nc * 16 + colb;
#pragma unroll
        for (int v = 0; v < 8; ++v) {
            int r = rowbase + rb + v;
            if (r < Nrows) {
                float o = acc[nc][v];
                if (addC)   o += addC[(size_t)r * Nout + col];
                if (doRelu) o = fmaxf(o, 0.f);
                Y[(size_t)r * Nout + col] = o;
            }
        }
    }
}

// ---------------------------------------------------------------------------
// GraphNorm helpers (per-feature stats over all N rows)
// ---------------------------------------------------------------------------
__global__ void zero_kernel(float* __restrict__ p, int n)
{
    int i = blockIdx.x * blockDim.x + threadIdx.x;
    if (i < n) p[i] = 0.f;
}

__global__ __launch_bounds__(256) void col_sum_kernel(
    const float* __restrict__ X, float* __restrict__ sums, int Nrows, int C)
{
    __shared__ float red[256];
    const int tid = threadIdx.x;
    const int c   = tid & (C - 1);
    const int rpb = 256 / C;
    float acc = 0.f;
    for (int r = blockIdx.x * rpb + tid / C; r < Nrows; r += gridDim.x * rpb)
        acc += X[(size_t)r * C + c];
    red[tid] = acc; __syncthreads();
    for (int off = 128; off >= C; off >>= 1) {
        if (tid < off) red[tid] += red[tid + off];
        __syncthreads();
    }
    if (tid < C) atomicAdd(&sums[c], red[tid]);
}

__global__ __launch_bounds__(256) void col_sqsum_kernel(
    const float* __restrict__ X, const float* __restrict__ sums,
    const float* __restrict__ ms, float* __restrict__ sumsq, int Nrows, int C)
{
    __shared__ float red[256];
    const int tid = threadIdx.x;
    const int c   = tid & (C - 1);
    const int rpb = 256 / C;
    const float mm = ms[c] * (sums[c] / (float)Nrows);
    float acc = 0.f;
    for (int r = blockIdx.x * rpb + tid / C; r < Nrows; r += gridDim.x * rpb) {
        float v = X[(size_t)r * C + c] - mm;
        acc += v * v;
    }
    red[tid] = acc; __syncthreads();
    for (int off = 128; off >= C; off >>= 1) {
        if (tid < off) red[tid] += red[tid + off];
        __syncthreads();
    }
    if (tid < C) atomicAdd(&sumsq[c], red[tid]);
}

__global__ void gn_apply_kernel(
    const float* __restrict__ X, const float* __restrict__ sums,
    const float* __restrict__ sumsq, const float* __restrict__ w,
    const float* __restrict__ b, const float* __restrict__ ms,
    const float* __restrict__ resid, float* __restrict__ Y, int Nrows, int C)
{
    int idx = blockIdx.x * blockDim.x + threadIdx.x;
    int total = Nrows * C;
    if (idx >= total) return;
    int c = idx % C;
    float mean = sums[c] / (float)Nrows;
    float var  = sumsq[c] / (float)Nrows;
    float v = X[idx] - ms[c] * mean;
    float o = v * rsqrtf(var + GN_EPS) * w[c] + b[c];
    o = fmaxf(o, 0.f);
    if (resid) o += resid[idx];
    Y[idx] = o;
}

// ---------------------------------------------------------------------------
// Orchestration
// ---------------------------------------------------------------------------
extern "C" void kernel_launch(void* const* d_in, const int* in_sizes, int n_in,
                              void* d_out, int out_size, void* d_ws, size_t ws_size,
                              hipStream_t stream)
{
    (void)n_in; (void)out_size; (void)ws_size;
    const float* x          = (const float*)d_in[0];
    const float* edge_attr  = (const float*)d_in[1];
    const int*   edge_index = (const int*)  d_in[2];
    const float* init_W     = (const float*)d_in[3];
    const float* init_b     = (const float*)d_in[4];
    const float* emW        = (const float*)d_in[5];   // [L,32,1024]
    const float* emb        = (const float*)d_in[6];   // [L,1024]
    const float* root_W     = (const float*)d_in[7];   // [L,32,32]
    const float* root_b     = (const float*)d_in[8];
    const float* gn_w       = (const float*)d_in[9];
    const float* gn_b       = (const float*)d_in[10];
    const float* gn_ms      = (const float*)d_in[11];
    const float* trans_W    = (const float*)d_in[12];  // [L,64,32]
    const float* trans_b    = (const float*)d_in[13];
    const float* final_W    = (const float*)d_in[14];  // [32,64]
    const float* final_b    = (const float*)d_in[15];
    const float* fgn_w      = (const float*)d_in[16];
    const float* fgn_b      = (const float*)d_in[17];
    const float* fgn_ms     = (const float*)d_in[18];

    const int N = in_sizes[0] / 64;   // 20000
    const int E = in_sizes[1] / 32;   // 80000
    const int H = 32, L = 3;

    float* ws    = (float*)d_ws;
    float* h0    = ws;
    float* h1    = h0 + (size_t)N * H;
    float* hc    = h1 + (size_t)N * H;
    float* agg   = hc + (size_t)N * H;
    float* sums  = agg + (size_t)N * H;
    float* sumsq = sums + 64;
    float* out   = (float*)d_out;

    const dim3 blk(128);
    const int gN = (N + 63) / 64;
    const int gE = (E + 63) / 64;

    // h = x @ init_W + init_b        (K=64, Nout=32)
    node_gemm_kernel<2, 2><<<gN, blk, 0, stream>>>(x, nullptr, init_W, init_b,
                                                   nullptr, h0, N, 0);

    float* hcur = h0;
    float* hnxt = h1;
    for (int i = 0; i < L; ++i) {
        zero_kernel<<<(N * H + 255) / 256, 256, 0, stream>>>(agg, N * H);
        // fused edge MLP + message + scatter-add
        edge_conv_kernel<<<gE, blk, 0, stream>>>(
            edge_attr, edge_index, hcur,
            emW + (size_t)i * 32 * 1024, emb + (size_t)i * 1024, agg, E, N);
        // conv = agg + h @ root_W + root_b   (in place into agg)
        node_gemm_kernel<1, 2><<<gN, blk, 0, stream>>>(
            hcur, nullptr, root_W + (size_t)i * 1024, root_b + i * 32,
            agg, agg, N, 0);
        // GraphNorm stats
        zero_kernel<<<1, 128, 0, stream>>>(sums, 128);
        col_sum_kernel<<<128, 256, 0, stream>>>(agg, sums, N, 32);
        col_sqsum_kernel<<<128, 256, 0, stream>>>(agg, sums, gn_ms + i * 32,
                                                  sumsq, N, 32);
        // hc = relu(GN(conv)) + h
        gn_apply_kernel<<<(N * 32 + 255) / 256, 256, 0, stream>>>(
            agg, sums, sumsq, gn_w + i * 32, gn_b + i * 32, gn_ms + i * 32,
            hcur, hc, N, 32);
        // h = relu(concat([h,hc]) @ trans_W + trans_b)
        node_gemm_kernel<2, 2><<<gN, blk, 0, stream>>>(
            hcur, hc, trans_W + (size_t)i * 2048, trans_b + i * 32,
            nullptr, hnxt, N, 1);
        float* t = hcur; hcur = hnxt; hnxt = t;
    }

    // out_pre = h @ final_W + final_b  (K=32, Nout=64), written to d_out
    node_gemm_kernel<1, 4><<<gN, blk, 0, stream>>>(hcur, nullptr, final_W,
                                                   final_b, nullptr, out, N, 0);
    // final GraphNorm + relu, in place on d_out
    zero_kernel<<<1, 128, 0, stream>>>(sums, 128);
    col_sum_kernel<<<128, 256, 0, stream>>>(out, sums, N, 64);
    col_sqsum_kernel<<<128, 256, 0, stream>>>(out, sums, fgn_ms, sumsq, N, 64);
    gn_apply_kernel<<<(N * 64 + 255) / 256, 256, 0, stream>>>(
        out, sums, sumsq, fgn_w, fgn_b, fgn_ms, nullptr, out, N, 64);
}
